// EpisodicAdapter_10264971837862
// MI455X (gfx1250) — compile-verified, matrix operand order checked by reading, not run on hardware
//
#include <hip/hip_runtime.h>
#include <hip/hip_bf16.h>

// ---------------------------------------------------------------------------
// CDNA5 (gfx1250) LoRA-attention block.
//   H=4096, NH=32, NKV=8, HD=128, R=16, B=4, Qlen=Tlen=1024, SCALING=2.0
// All heavy math runs through v_wmma_f32_16x16x32_bf16 (wave32).
// ---------------------------------------------------------------------------

typedef __attribute__((ext_vector_type(16))) __bf16 bf16x16;
typedef __attribute__((ext_vector_type(8)))  float  f32x8;

__device__ __forceinline__ f32x8 zero8() {
    f32x8 z;
#pragma unroll
    for (int e = 0; e < 8; ++e) z[e] = 0.0f;
    return z;
}

// Load a 16x32 bf16 fragment (A: MxK row-major tile, or B: NxK "B-transposed"
// tile) from LDS, per ISA 7.12.2 16-bit A-matrix layout:
//   lane l -> row l%16 ; K-halves 0-7 (lanes 0-15) / 8-15 (lanes 16-31) in
//   VGPRs 0-3, +16 in VGPRs 4-7.  Element e of the v16bf maps to K:
//   e in [0,8)  -> k = hi + e          (hi = (lane>=16) ? 8 : 0)
//   e in [8,16) -> k = 16 + hi + e-8
__device__ __forceinline__ bf16x16 load_frag(const __bf16* tile, int stride, int lane) {
    const int r  = lane & 15;
    const int hi = (lane & 16) ? 8 : 0;
    const __bf16* p = tile + r * stride;
    bf16x16 f;
#pragma unroll
    for (int e = 0; e < 8; ++e) f[e] = p[hi + e];
#pragma unroll
    for (int e = 0; e < 8; ++e) f[8 + e] = p[16 + hi + e];
    return f;
}

__device__ __forceinline__ f32x8 wmma_bf16(bf16x16 a, bf16x16 b, f32x8 c) {
    // 8 args: (neg_a, A, neg_b, B, c_mod, C, reuse_a, reuse_b)
    return __builtin_amdgcn_wmma_f32_16x16x32_bf16(false, a, false, b,
                                                   (short)0, c, false, false);
}

// ---------------------------------------------------------------------------
// xA = scale * (X @ A),  X: 4096x4096 (f32 or bf16), A: 4096x16, out: 4096x16
// One block per row; 16 chunks x 16 cols.
// ---------------------------------------------------------------------------
__global__ __launch_bounds__(256)
void ea_xa_kernel(const void* __restrict__ X, int x_is_bf16,
                  const float* __restrict__ A, float* __restrict__ out,
                  float scale)
{
    __shared__ float red[256];
    const int row = blockIdx.x;
    const int r   = threadIdx.x & 15;
    const int ch  = threadIdx.x >> 4;
    const int k0  = ch * 256;
    float s = 0.0f;
    if (x_is_bf16) {
        const __bf16* x = (const __bf16*)X + (size_t)row * 4096 + k0;
#pragma unroll 8
        for (int k = 0; k < 256; ++k) s += (float)x[k] * A[(size_t)(k0 + k) * 16 + r];
    } else {
        const float* x = (const float*)X + (size_t)row * 4096 + k0;
#pragma unroll 8
        for (int k = 0; k < 256; ++k) s += x[k] * A[(size_t)(k0 + k) * 16 + r];
    }
    red[threadIdx.x] = s;
    __syncthreads();
    if (ch == 0) {
        float t = 0.0f;
#pragma unroll
        for (int c = 0; c < 16; ++c) t += red[c * 16 + r];
        out[(size_t)row * 16 + r] = scale * t;
    }
}

// ---------------------------------------------------------------------------
// C = cvt_bf16(X) @ cvt_bf16(W)  +  xA @ Blora   (LoRA folded as extra K tile)
// X: 4096 x 4096 (f32 or bf16), W: 4096 x N, xA: 4096 x 16 (pre-scaled),
// Blora: 16 x N.  Block tile 128x128, 8 waves each own 32x64 (2x4 WMMA tiles).
// ---------------------------------------------------------------------------
__global__ __launch_bounds__(256)
void ea_gemm_lora_kernel(const void* __restrict__ Aptr, int a_is_bf16,
                         const float* __restrict__ W,
                         const float* __restrict__ xA,
                         const float* __restrict__ Bl,
                         void* __restrict__ Cptr, int c_is_bf16,
                         const float* __restrict__ resid,
                         int N)
{
    __shared__ __bf16 As[128 * 34];   // [m][k], stride 34 kills bank conflicts
    __shared__ __bf16 Bs[128 * 34];   // [n][k]
    const int tid  = threadIdx.x;
    const int lane = tid & 31, wid = tid >> 5;
    const int wr = wid & 3, wc = wid >> 2;          // 4x2 wave grid
    const int bm = blockIdx.x * 128, bn = blockIdx.y * 128;

    f32x8 acc[2][4];
#pragma unroll
    for (int i = 0; i < 2; ++i)
#pragma unroll
        for (int j = 0; j < 4; ++j) acc[i][j] = zero8();

    const int nkt = 4096 / 32;                       // 128 main tiles + 1 LoRA
    for (int kt = 0; kt <= nkt; ++kt) {
        // ---- stage A tile (128 rows x 32 k) into LDS as bf16 ----
        {
            const int r  = tid >> 1;
            const int c0 = (tid & 1) * 16;
            __bf16* dst = &As[r * 34 + c0];
            if (kt < nkt) {
                const int gk = kt * 32 + c0;
                if (a_is_bf16) {
                    const __bf16* src = (const __bf16*)Aptr + (size_t)(bm + r) * 4096 + gk;
#pragma unroll
                    for (int i = 0; i < 16; ++i) dst[i] = src[i];
                } else {
                    const float* src = (const float*)Aptr + (size_t)(bm + r) * 4096 + gk;
#pragma unroll
                    for (int i = 0; i < 16; ++i) dst[i] = (__bf16)src[i];
                }
            } else {                                  // LoRA tile: [xA | 0]
                if (c0 == 0) {
                    const float* src = xA + (size_t)(bm + r) * 16;
#pragma unroll
                    for (int i = 0; i < 16; ++i) dst[i] = (__bf16)src[i];
                } else {
#pragma unroll
                    for (int i = 0; i < 16; ++i) dst[i] = (__bf16)0.0f;
                }
            }
        }
        // ---- stage B tile (32 k x 128 n) transposed into LDS [n][k] ----
        {
            const int k  = tid >> 3;
            const int n0 = (tid & 7) * 16;
            if (kt < nkt) {
                const float* src = W + (size_t)(kt * 32 + k) * N + bn + n0;
                if (kt + 1 < nkt)
                    __builtin_prefetch(src + (size_t)32 * N, 0, 0); // global_prefetch_b8
#pragma unroll
                for (int i = 0; i < 16; ++i) Bs[(n0 + i) * 34 + k] = (__bf16)src[i];
            } else if (k < 16) {                      // LoRA tile: [Blora ; 0]
                const float* src = Bl + (size_t)k * N + bn + n0;
#pragma unroll
                for (int i = 0; i < 16; ++i) Bs[(n0 + i) * 34 + k] = (__bf16)src[i];
            } else {
#pragma unroll
                for (int i = 0; i < 16; ++i) Bs[(n0 + i) * 34 + k] = (__bf16)0.0f;
            }
        }
        __syncthreads();

        bf16x16 af[2], bfr[4];
#pragma unroll
        for (int i = 0; i < 2; ++i)
            af[i] = load_frag(&As[(wr * 32 + 16 * i) * 34], 34, lane);
#pragma unroll
        for (int j = 0; j < 4; ++j)
            bfr[j] = load_frag(&Bs[(wc * 64 + 16 * j) * 34], 34, lane);
#pragma unroll
        for (int i = 0; i < 2; ++i)
#pragma unroll
            for (int j = 0; j < 4; ++j)
                acc[i][j] = wmma_bf16(af[i], bfr[j], acc[i][j]);
        __syncthreads();
    }

    // ---- epilogue (D layout: lane -> col lane%16, vgpr v -> row v + 8*hi) ----
    const int hi8 = (lane & 16) ? 8 : 0;
    const int cn  = lane & 15;
#pragma unroll
    for (int i = 0; i < 2; ++i)
#pragma unroll
        for (int j = 0; j < 4; ++j)
#pragma unroll
            for (int v = 0; v < 8; ++v) {
                const size_t gm = (size_t)bm + wr * 32 + 16 * i + v + hi8;
                const size_t gn = (size_t)bn + wc * 64 + 16 * j + cn;
                const float val = acc[i][j][v];
                if (c_is_bf16) {
                    ((__bf16*)Cptr)[gm * N + gn] = (__bf16)val;
                } else {
                    const float rr = resid ? resid[gm * N + gn] : 0.0f;
                    ((float*)Cptr)[gm * N + gn] = val + rr;
                }
            }
}

// ---------------------------------------------------------------------------
// Flash attention, GQA (32 q heads / 8 kv heads, HD=128).
// Grid: (B*NH, Qlen/64).  Block: 128 threads = 4 waves; each wave owns 16 q
// rows. T processed in chunks of 64 with online softmax.  All matmuls WMMA.
// ---------------------------------------------------------------------------
__global__ __launch_bounds__(128)
void ea_attn_kernel(const __bf16* __restrict__ Q,   // (B*1024) x 4096
                    const __bf16* __restrict__ K,   // (B*1024) x 1024
                    const __bf16* __restrict__ V,   // (B*1024) x 1024
                    __bf16* __restrict__ ctx)       // (B*1024) x 4096
{
    __shared__ __bf16 Qs[64 * 130];   // [q][d]
    __shared__ __bf16 Ks[64 * 130];   // [t][d]
    __shared__ __bf16 Vt[128 * 66];   // [d][t] (transposed -> B-frag layout)
    __shared__ __bf16 Ps[64 * 66];    // [q][t] (D-layout -> A-layout bounce)

    const int tid = threadIdx.x, lane = tid & 31, wid = tid >> 5;
    const int bh = blockIdx.x;                 // b*32 + h
    const int qt = blockIdx.y;                 // q tile (64 rows)
    const int b = bh >> 5, h = bh & 31;
    const int kvh = h >> 2;                    // GQA: 4 q heads per kv head
    const int qb = wid * 16;
    const int hi8 = (lane & 16) ? 8 : 0;
    const int cn  = lane & 15;

    { // load Q tile (64 x 128)
        const int r = tid >> 1, d0 = (tid & 1) * 64;
        const __bf16* src = Q + (size_t)(b * 1024 + qt * 64 + r) * 4096 + h * 128 + d0;
        __bf16* dst = &Qs[r * 130 + d0];
#pragma unroll 8
        for (int i = 0; i < 64; ++i) dst[i] = src[i];
    }

    float m_row[8], l_row[8];
    f32x8 o[8];
#pragma unroll
    for (int v = 0; v < 8; ++v) { m_row[v] = -1e30f; l_row[v] = 0.0f; }
#pragma unroll
    for (int dt = 0; dt < 8; ++dt) o[dt] = zero8();

    const float sc = 0.08838834764831845f;     // 1/sqrt(128)

    for (int tc = 0; tc < 16; ++tc) {
        const int t0 = tc * 64;
        { // K tile (64 x 128) row-major
            const int r = tid >> 1, d0 = (tid & 1) * 64;
            const __bf16* src = K + (size_t)(b * 1024 + t0 + r) * 1024 + kvh * 128 + d0;
            __bf16* dst = &Ks[r * 130 + d0];
#pragma unroll 8
            for (int i = 0; i < 64; ++i) dst[i] = src[i];
        }
        { // V tile transposed into [d][t]
            const int r = tid >> 1, d0 = (tid & 1) * 64;
            const __bf16* src = V + (size_t)(b * 1024 + t0 + r) * 1024 + kvh * 128 + d0;
#pragma unroll 8
            for (int i = 0; i < 64; ++i) Vt[(d0 + i) * 66 + r] = src[i];
        }
        __syncthreads();

        // S = Q @ K^T  (wave: 16q x 64t -> 4 tiles, K-dim = d = 4 x 32)
        f32x8 s[4];
#pragma unroll
        for (int j = 0; j < 4; ++j) s[j] = zero8();
#pragma unroll
        for (int kk = 0; kk < 4; ++kk) {
            bf16x16 a = load_frag(&Qs[qb * 130 + kk * 32], 130, lane);
#pragma unroll
            for (int j = 0; j < 4; ++j) {
                bf16x16 bb = load_frag(&Ks[(j * 16) * 130 + kk * 32], 130, lane);
                s[j] = wmma_bf16(a, bb, s[j]);
            }
        }

        // online softmax (row stats across 16-lane groups of the D layout)
        float mx[8];
#pragma unroll
        for (int v = 0; v < 8; ++v) mx[v] = -1e30f;
#pragma unroll
        for (int j = 0; j < 4; ++j)
#pragma unroll
            for (int v = 0; v < 8; ++v) {
                s[j][v] *= sc;
                mx[v] = fmaxf(mx[v], s[j][v]);
            }
#pragma unroll
        for (int m = 1; m < 16; m <<= 1)
#pragma unroll
            for (int v = 0; v < 8; ++v)
                mx[v] = fmaxf(mx[v], __shfl_xor(mx[v], m, 32));

        float alpha[8], rs[8];
#pragma unroll
        for (int v = 0; v < 8; ++v) {
            const float mn = fmaxf(m_row[v], mx[v]);
            alpha[v] = __expf(m_row[v] - mn);
            m_row[v] = mn;
            rs[v] = 0.0f;
        }
#pragma unroll
        for (int j = 0; j < 4; ++j)
#pragma unroll
            for (int v = 0; v < 8; ++v) {
                const float p = __expf(s[j][v] - m_row[v]);
                s[j][v] = p;
                rs[v] += p;
            }
#pragma unroll
        for (int m = 1; m < 16; m <<= 1)
#pragma unroll
            for (int v = 0; v < 8; ++v)
                rs[v] += __shfl_xor(rs[v], m, 32);
#pragma unroll
        for (int v = 0; v < 8; ++v) l_row[v] = l_row[v] * alpha[v] + rs[v];
#pragma unroll
        for (int dt = 0; dt < 8; ++dt)
#pragma unroll
            for (int v = 0; v < 8; ++v) o[dt][v] *= alpha[v];

        // bounce P through LDS (D-layout -> A-layout), bf16
#pragma unroll
        for (int j = 0; j < 4; ++j)
#pragma unroll
            for (int v = 0; v < 8; ++v)
                Ps[(qb + v + hi8) * 66 + j * 16 + cn] = (__bf16)s[j][v];
        __syncthreads();

        // O += P @ V   (k = t = 2 x 32, n = d = 8 tiles)
#pragma unroll
        for (int kk = 0; kk < 2; ++kk) {
            bf16x16 a = load_frag(&Ps[qb * 66 + kk * 32], 66, lane);
#pragma unroll
            for (int dt = 0; dt < 8; ++dt) {
                bf16x16 bb = load_frag(&Vt[(dt * 16) * 66 + kk * 32], 66, lane);
                o[dt] = wmma_bf16(a, bb, o[dt]);
            }
        }
        __syncthreads();
    }

    // epilogue: O /= l, write bf16 context
#pragma unroll
    for (int dt = 0; dt < 8; ++dt)
#pragma unroll
        for (int v = 0; v < 8; ++v) {
            const int q = qt * 64 + qb + v + hi8;
            const int d = dt * 16 + cn;
            const float val = o[dt][v] / l_row[v];
            ctx[(size_t)(b * 1024 + q) * 4096 + h * 128 + d] = (__bf16)val;
        }
}

// ---------------------------------------------------------------------------
extern "C" void kernel_launch(void* const* d_in, const int* in_sizes, int n_in,
                              void* d_out, int out_size, void* d_ws, size_t ws_size,
                              hipStream_t stream)
{
    (void)in_sizes; (void)n_in; (void)out_size; (void)ws_size;
    const float* hidden = (const float*)d_in[0];
    const float* traces = (const float*)d_in[1];
    const float* Wq = (const float*)d_in[2];
    const float* Aq = (const float*)d_in[3];
    const float* Bq = (const float*)d_in[4];
    const float* Wk = (const float*)d_in[5];
    const float* Ak = (const float*)d_in[6];
    const float* Bk = (const float*)d_in[7];
    const float* Wv = (const float*)d_in[8];
    const float* Av = (const float*)d_in[9];
    const float* Bv = (const float*)d_in[10];
    const float* Wo = (const float*)d_in[11];
    const float* Ao = (const float*)d_in[12];
    const float* Bo = (const float*)d_in[13];
    float* out = (float*)d_out;

    // workspace carve-up (~85 MB)
    char* ws = (char*)d_ws;
    __bf16* Qbf = (__bf16*)ws;  ws += (size_t)4096 * 4096 * 2;
    __bf16* Kbf = (__bf16*)ws;  ws += (size_t)4096 * 1024 * 2;
    __bf16* Vbf = (__bf16*)ws;  ws += (size_t)4096 * 1024 * 2;
    __bf16* Ctx = (__bf16*)ws;  ws += (size_t)4096 * 4096 * 2;
    float* xAq = (float*)ws;    ws += (size_t)4096 * 16 * 4;
    float* xAk = (float*)ws;    ws += (size_t)4096 * 16 * 4;
    float* xAv = (float*)ws;    ws += (size_t)4096 * 16 * 4;
    float* xAo = (float*)ws;    ws += (size_t)4096 * 16 * 4;

    const float scaling = 2.0f;   // lora_alpha / lora_r = 32/16

    // LoRA down-projections
    ea_xa_kernel<<<4096, 256, 0, stream>>>(hidden, 0, Aq, xAq, scaling);
    ea_xa_kernel<<<4096, 256, 0, stream>>>(traces, 0, Ak, xAk, scaling);
    ea_xa_kernel<<<4096, 256, 0, stream>>>(traces, 0, Av, xAv, scaling);

    // Q/K/V projections (LoRA folded as extra K-tile), bf16 outputs
    ea_gemm_lora_kernel<<<dim3(32, 32), 256, 0, stream>>>(hidden, 0, Wq, xAq, Bq,
                                                          Qbf, 1, nullptr, 4096);
    ea_gemm_lora_kernel<<<dim3(32, 8), 256, 0, stream>>>(traces, 0, Wk, xAk, Bk,
                                                         Kbf, 1, nullptr, 1024);
    ea_gemm_lora_kernel<<<dim3(32, 8), 256, 0, stream>>>(traces, 0, Wv, xAv, Bv,
                                                         Vbf, 1, nullptr, 1024);

    // GQA flash attention -> bf16 context
    ea_attn_kernel<<<dim3(128, 16), 128, 0, stream>>>(Qbf, Kbf, Vbf, Ctx);

    // output projection with LoRA + residual, f32 output
    ea_xa_kernel<<<4096, 256, 0, stream>>>(Ctx, 1, Ao, xAo, scaling);
    ea_gemm_lora_kernel<<<dim3(32, 32), 256, 0, stream>>>(Ctx, 1, Wo, xAo, Bo,
                                                          out, 0, hidden, 4096);
}
